// VCLinear_40656160424049
// MI455X (gfx1250) — matrix operand order repaired and checked by reading, not compiled, not run on hardware
//
#include <hip/hip_runtime.h>

// ---------------------------------------------------------------------------
// Varying-coefficient linear layer:
//   out = Xaug(2048x4096) @ Wflat(4096x512) + basis(2048x8) @ b_wts(8x512)
//
// Fast path (ws_size >= 20MB):
//   1) prep_xaug : Xaug bf16 (basis scale fused)         -> ws   [16 MB]
//   2) prep_wt   : W  f32 -> bf16, transposed to N-major -> ws   [ 4 MB]
//   3) gemm      : double-buffered TDM (tensor_load_to_lds) tile staging,
//                  overlapped with v_wmma bf16 compute (TENSORcnt throttle)
// Fallback: single fused kernel converting f32->bf16 in-loop.
// ---------------------------------------------------------------------------

typedef __attribute__((ext_vector_type(16))) __bf16 bf16x16;
typedef __attribute__((ext_vector_type(8)))  float  floatx8;
typedef __attribute__((ext_vector_type(4)))  unsigned int u32x4;
typedef __attribute__((ext_vector_type(8)))  int   i32x8;
typedef __attribute__((ext_vector_type(4)))  int   i32x4;
typedef unsigned short u16;

constexpr int BATCH   = 2048;
constexpr int IN_DIM  = 512;
constexpr int OUT_DIM = 512;
constexpr int SDIM    = 8;
constexpr int KDIM    = SDIM * IN_DIM;   // 4096

constexpr int MT = 64;
constexpr int NT = 128;
constexpr int KT = 64;
constexpr int THREADS = 256;             // 8 waves (wave32)
constexpr int LDS_STRIDE = KT + 8;       // 72 bf16 = 144 B rows (16B aligned)
constexpr int KITERS = KDIM / KT;        // 64

#if defined(__has_builtin)
#if __has_builtin(__builtin_amdgcn_tensor_load_to_lds)
#define HAVE_TDM 1
#endif
#endif
#ifndef HAVE_TDM
#define HAVE_TDM 0
#endif

// ------------------------------ helpers -----------------------------------

__device__ __forceinline__ u16 f32_to_bf16_bits(float x) {
    union { __bf16 h; u16 u; } cv;
    cv.h = (__bf16)x;          // v_cvt RNE
    return cv.u;
}

__device__ __forceinline__ float basis_val(int s, float tv) {
    // [1, t, t^2, t^3, relu(t-0.2)^3, relu(t-0.4)^3, relu(t-0.6)^3, relu(t-0.8)^3]
    float t2 = tv * tv;
    switch (s) {
        case 0: return 1.0f;
        case 1: return tv;
        case 2: return t2;
        case 3: return t2 * tv;
        default: {
            float r = tv - 0.2f * (float)(s - 3);
            r = r > 0.0f ? r : 0.0f;
            return r * r * r;
        }
    }
}

#if HAVE_TDM
// Issue a 2D TDM load: tile (tile_d0 x tile_d1) of bf16 from a row-major
// tensor (dim0 contiguous, stride_d0 elements between rows) into LDS at
// lds_byte_off. LDS padding: +4 DWORDs after each 32 DWORDs (=> 64 bf16 rows
// land at a 72-bf16 stride, matching LDS_STRIDE). tile_d0 must be 64.
__device__ __forceinline__ void tdm_load_2d(unsigned int lds_byte_off,
                                            const void* gaddr,
                                            unsigned int tensor_d0,
                                            unsigned int tensor_d1,
                                            unsigned int tile_d0,
                                            unsigned int tile_d1,
                                            unsigned int stride_d0) {
    unsigned long long ga = (unsigned long long)(uintptr_t)gaddr;
    u32x4 g0;
    g0.x = 1u;                                            // count=1 (valid), user desc
    g0.y = lds_byte_off;                                  // lds_addr
    g0.z = (unsigned int)ga;                              // global_addr[31:0]
    g0.w = ((unsigned int)(ga >> 32) & 0x01FFFFFFu)       // global_addr[56:32]
           | 0x80000000u;                                 // type=2 ("image")
    i32x8 g1;
    g1[0] = (int)((1u << 16)      // data_size = 1 -> 2 bytes
                | (1u << 20)      // pad_enable
                | (4u << 22)      // pad_interval: every 32 DWORDs (128 B row)
                | (3u << 25));    // pad_amount: 4 DWORDs (16 B)
    g1[1] = (int)((tensor_d0 & 0xFFFFu) << 16);                           // abar=0 | td0.lo
    g1[2] = (int)(((tensor_d0 >> 16) & 0xFFFFu) | ((tensor_d1 & 0xFFFFu) << 16));
    g1[3] = (int)(((tensor_d1 >> 16) & 0xFFFFu) | (tile_d0 << 16));
    g1[4] = (int)(tile_d1 & 0xFFFFu);                                     // tile_dim2=0
    g1[5] = (int)stride_d0;                                               // dim0 stride lo
    g1[6] = 0;                                                            // stride hi / dim1 stride
    g1[7] = 0;
    i32x4 z4 = {0, 0, 0, 0};
    i32x8 z8 = {0, 0, 0, 0, 0, 0, 0, 0};
    // 6-arg form (clang-23 / therock headers): (g0, g1, g2, g3, g4, cpol)
    __builtin_amdgcn_tensor_load_to_lds(g0, g1, z4, z4, z8, 0);
}
#endif

// --------------------------- prep kernels ---------------------------------

// Xaug[b, k] = basis(t_b)[k>>9] * features[b, k&511], bf16. One block per row.
__global__ __launch_bounds__(THREADS)
void prep_xaug(const float* __restrict__ inputs, u16* __restrict__ Xaug) {
    const int row = blockIdx.x;
    const int t   = threadIdx.x;
    const float tv = inputs[(size_t)row * (IN_DIM + 1)];
    const int k0   = t * 16;                 // 16 contiguous k per thread
    const float bs = basis_val(k0 >> 9, tv); // 16 | 512 -> single s per thread
    const float* src = inputs + (size_t)row * (IN_DIM + 1) + 1 + (k0 & (IN_DIM - 1));
    union { u16 us[16]; uint4 u4[2]; } pk;
    #pragma unroll
    for (int j = 0; j < 16; ++j) pk.us[j] = f32_to_bf16_bits(bs * src[j]);
    uint4* dst = (uint4*)(Xaug + (size_t)row * KDIM + k0);
    dst[0] = pk.u4[0];
    dst[1] = pk.u4[1];
}

// WT[n, k] = bf16(W[k, n]) : LDS-tiled 32x32 transpose. grid (4096/32, 512/32)
__global__ __launch_bounds__(THREADS)
void prep_wt(const float* __restrict__ W, u16* __restrict__ WT) {
    __shared__ u16 tile[32][33];
    const int kb = blockIdx.x * 32;
    const int nb = blockIdx.y * 32;
    const int t  = threadIdx.x;
    {
        const int lk = t >> 3, ln0 = (t & 7) * 4;
        const float* src = W + (size_t)(kb + lk) * OUT_DIM + nb + ln0;
        #pragma unroll
        for (int j = 0; j < 4; ++j) tile[lk][ln0 + j] = f32_to_bf16_bits(src[j]);
    }
    __syncthreads();
    {
        const int ln = t >> 3, lk0 = (t & 7) * 4;
        u16* dst = WT + (size_t)(nb + ln) * KDIM + kb + lk0;
        #pragma unroll
        for (int j = 0; j < 4; ++j) dst[j] = tile[lk0 + j][ln];
    }
}

// ------------------------- fast-path GEMM kernel ---------------------------

__global__ __launch_bounds__(THREADS)
void vclinear_wmma_pre(const float* __restrict__ inputs,
                       const float* __restrict__ bwts,
                       const u16* __restrict__ Xaug,   // (2048, 4096) bf16
                       const u16* __restrict__ WT,     // (512, 4096)  bf16, N-major
                       float* __restrict__ out) {
    __shared__ float basisLds[MT][SDIM];
    __shared__ float bwLds[SDIM][NT];
    __shared__ __align__(16) u16 Alds[2][MT][LDS_STRIDE];   // double-buffered
    __shared__ __align__(16) u16 Blds[2][NT][LDS_STRIDE];

    const int t       = threadIdx.x;
    const int rowBase = blockIdx.y * MT;
    const int colBase = blockIdx.x * NT;

    if (t < MT) {
        float tv = inputs[(size_t)(rowBase + t) * (IN_DIM + 1)];
        #pragma unroll
        for (int s = 0; s < SDIM; ++s) basisLds[t][s] = basis_val(s, tv);
    }
    {
        int s  = t >> 5;
        int n0 = (t & 31) * (NT / 32);
        #pragma unroll
        for (int j = 0; j < NT / 32; ++j)
            bwLds[s][n0 + j] = bwts[(size_t)s * OUT_DIM + colBase + n0 + j];
    }
    __syncthreads();

#if HAVE_TDM
    const unsigned int aOff0 = (unsigned int)(uintptr_t)&Alds[0][0][0];
    const unsigned int aOff1 = (unsigned int)(uintptr_t)&Alds[1][0][0];
    const unsigned int bOff0 = (unsigned int)(uintptr_t)&Blds[0][0][0];
    const unsigned int bOff1 = (unsigned int)(uintptr_t)&Blds[1][0][0];
    // Prologue: kick tile 0 into buffer 0 while we do the bias init below.
    // TDM ignores EXEC -> gate with a readfirstlane-uniform scalar branch.
    if (__builtin_amdgcn_readfirstlane((int)(threadIdx.x >> 5)) == 0) {
        tdm_load_2d(aOff0, Xaug + (size_t)rowBase * KDIM, KDIM, BATCH,  KT, MT, KDIM);
        tdm_load_2d(bOff0, WT   + (size_t)colBase * KDIM, KDIM, OUT_DIM, KT, NT, KDIM);
    }
#endif

    const int wave  = t >> 5;
    const int lane  = t & 31;
    const int laneM = lane & 15;
    const int h     = lane >> 4;
    const int wm    = wave & 1;
    const int wn    = wave >> 1;

    // bias init (exact f32): basis @ b_wts  (overlaps with TDM tile 0)
    floatx8 acc[2][2];
    #pragma unroll
    for (int mt = 0; mt < 2; ++mt)
        #pragma unroll
        for (int nt = 0; nt < 2; ++nt) {
            const int N = wn * 32 + nt * 16 + laneM;
            #pragma unroll
            for (int r = 0; r < 8; ++r) {
                const int M = wm * 32 + mt * 16 + h * 8 + r;
                float v = 0.0f;
                #pragma unroll
                for (int s = 0; s < SDIM; ++s) v += basisLds[M][s] * bwLds[s][N];
                acc[mt][nt][r] = v;
            }
        }

    for (int kb = 0; kb < KITERS; ++kb) {
        const int buf = kb & 1;
        __syncthreads();   // buffer (kb+1)&1 no longer being read by any wave

#if HAVE_TDM
        if (__builtin_amdgcn_readfirstlane((int)(threadIdx.x >> 5)) == 0) {
            if (kb + 1 < KITERS) {
                const int iK = (kb + 1) * KT;
                tdm_load_2d(buf ? aOff0 : aOff1,
                            Xaug + (size_t)rowBase * KDIM + iK,
                            KDIM, BATCH, KT, MT, KDIM);
                tdm_load_2d(buf ? bOff0 : bOff1,
                            WT + (size_t)colBase * KDIM + iK,
                            KDIM, OUT_DIM, KT, NT, KDIM);
                // TDM completes in order per wave: <=2 outstanding means the
                // pair for tile kb is done; tile kb+1 streams in background.
                __builtin_amdgcn_s_wait_tensorcnt(2);
            } else {
                __builtin_amdgcn_s_wait_tensorcnt(0);
            }
        }
#else
        {   // A tile kb: 64x64 bf16, 2x b128 per thread
            const int iK = kb * KT;
            const int row = t >> 2, c0 = (t & 3) * 16;
            const uint4* src = (const uint4*)(Xaug + (size_t)(rowBase + row) * KDIM + iK + c0);
            uint4* dst = (uint4*)&Alds[buf][row][c0];
            dst[0] = src[0]; dst[1] = src[1];
        }
        {   // B tile kb: 128x64 bf16, 4x b128 per thread (contiguous: WT is N-major)
            const int iK = kb * KT;
            const int n = t >> 1, c0 = (t & 1) * 32;
            const uint4* src = (const uint4*)(WT + (size_t)(colBase + n) * KDIM + iK + c0);
            uint4* dst = (uint4*)&Blds[buf][n][c0];
            dst[0] = src[0]; dst[1] = src[1]; dst[2] = src[2]; dst[3] = src[3];
        }
#endif
        __syncthreads();   // tile kb resident in buffer `buf`

        #pragma unroll
        for (int ks = 0; ks < 2; ++ks) {
            const int k0 = ks * 32;
            bf16x16 afrag[2], bfrag[2];
            #pragma unroll
            for (int mt = 0; mt < 2; ++mt) {
                const u16* arow = &Alds[buf][wm * 32 + mt * 16 + laneM][k0];
                union { uint4 u[2]; bf16x16 v; } ua;
                ua.u[0] = *(const uint4*)(arow + h * 8);
                ua.u[1] = *(const uint4*)(arow + 16 + h * 8);
                afrag[mt] = ua.v;
            }
            #pragma unroll
            for (int nt = 0; nt < 2; ++nt) {
                const u16* brow = &Blds[buf][wn * 32 + nt * 16 + laneM][k0];
                union { uint4 u[2]; bf16x16 v; } ub;
                ub.u[0] = *(const uint4*)(brow + h * 8);
                ub.u[1] = *(const uint4*)(brow + 16 + h * 8);
                bfrag[nt] = ub.v;
            }
            #pragma unroll
            for (int mt = 0; mt < 2; ++mt)
                #pragma unroll
                for (int nt = 0; nt < 2; ++nt)
                    acc[mt][nt] = __builtin_amdgcn_wmma_f32_16x16x32_bf16(
                        false, afrag[mt], false, bfrag[nt],
                        (short)0, acc[mt][nt], false, false);
        }
    }

    #pragma unroll
    for (int mt = 0; mt < 2; ++mt)
        #pragma unroll
        for (int nt = 0; nt < 2; ++nt) {
            const int N = colBase + wn * 32 + nt * 16 + laneM;
            #pragma unroll
            for (int r = 0; r < 8; ++r) {
                const int M = rowBase + wm * 32 + mt * 16 + h * 8 + r;
                out[(size_t)M * OUT_DIM + N] = acc[mt][nt][r];
            }
        }
}

// --------------------- fallback fused kernel (ws too small) -----------------

__global__ __launch_bounds__(THREADS)
void vclinear_wmma_fused(const float* __restrict__ inputs,
                         const float* __restrict__ W,
                         const float* __restrict__ bwts,
                         float* __restrict__ out) {
    __shared__ float basisLds[MT][SDIM];
    __shared__ float bwLds[SDIM][NT];
    __shared__ __align__(16) __bf16 Alds[MT][LDS_STRIDE];
    __shared__ __align__(16) __bf16 Blds[NT][LDS_STRIDE];

    const int t       = threadIdx.x;
    const int rowBase = blockIdx.y * MT;
    const int colBase = blockIdx.x * NT;

    if (t < MT) {
        float tv = inputs[(size_t)(rowBase + t) * (IN_DIM + 1)];
        #pragma unroll
        for (int s = 0; s < SDIM; ++s) basisLds[t][s] = basis_val(s, tv);
    }
    {
        int s  = t >> 5;
        int n0 = (t & 31) * (NT / 32);
        #pragma unroll
        for (int j = 0; j < NT / 32; ++j)
            bwLds[s][n0 + j] = bwts[(size_t)s * OUT_DIM + colBase + n0 + j];
    }
    __syncthreads();

    const int wave  = t >> 5;
    const int lane  = t & 31;
    const int laneM = lane & 15;
    const int h     = lane >> 4;
    const int wm    = wave & 1;
    const int wn    = wave >> 1;

    floatx8 acc[2][2];
    #pragma unroll
    for (int mt = 0; mt < 2; ++mt)
        #pragma unroll
        for (int nt = 0; nt < 2; ++nt) {
            const int N = wn * 32 + nt * 16 + laneM;
            #pragma unroll
            for (int r = 0; r < 8; ++r) {
                const int M = wm * 32 + mt * 16 + h * 8 + r;
                float v = 0.0f;
                #pragma unroll
                for (int s = 0; s < SDIM; ++s) v += basisLds[M][s] * bwLds[s][N];
                acc[mt][nt][r] = v;
            }
        }

    for (int kb = 0; kb < KITERS; ++kb) {
        const int s     = kb >> 3;
        const int iBase = (kb & 7) * KT;
        __syncthreads();
        {
            const int   row    = t >> 2;
            const int   c0     = (t & 3) * 16;
            const float bscale = basisLds[row][s];
            const float* src = inputs + (size_t)(rowBase + row) * (IN_DIM + 1) + 1 + iBase + c0;
            #pragma unroll
            for (int j = 0; j < 16; ++j) Alds[row][c0 + j] = (__bf16)(bscale * src[j]);
        }
        {
            const int kk = t >> 2;
            const int n0 = (t & 3) * 32;
            const float* src = W + (size_t)(s * IN_DIM + iBase + kk) * OUT_DIM + colBase + n0;
            #pragma unroll
            for (int j = 0; j < 32; ++j) Blds[n0 + j][kk] = (__bf16)src[j];
        }
        __syncthreads();

        #pragma unroll
        for (int ks = 0; ks < 2; ++ks) {
            const int k0 = ks * 32;
            bf16x16 afrag[2], bfrag[2];
            #pragma unroll
            for (int mt = 0; mt < 2; ++mt) {
                const __bf16* arow = &Alds[wm * 32 + mt * 16 + laneM][k0];
                union { uint4 u[2]; bf16x16 v; } ua;
                ua.u[0] = *(const uint4*)(arow + h * 8);
                ua.u[1] = *(const uint4*)(arow + 16 + h * 8);
                afrag[mt] = ua.v;
            }
            #pragma unroll
            for (int nt = 0; nt < 2; ++nt) {
                const __bf16* brow = &Blds[wn * 32 + nt * 16 + laneM][k0];
                union { uint4 u[2]; bf16x16 v; } ub;
                ub.u[0] = *(const uint4*)(brow + h * 8);
                ub.u[1] = *(const uint4*)(brow + 16 + h * 8);
                bfrag[nt] = ub.v;
            }
            #pragma unroll
            for (int mt = 0; mt < 2; ++mt)
                #pragma unroll
                for (int nt = 0; nt < 2; ++nt)
                    acc[mt][nt] = __builtin_amdgcn_wmma_f32_16x16x32_bf16(
                        false, afrag[mt], false, bfrag[nt],
                        (short)0, acc[mt][nt], false, false);
        }
    }

    #pragma unroll
    for (int mt = 0; mt < 2; ++mt)
        #pragma unroll
        for (int nt = 0; nt < 2; ++nt) {
            const int N = colBase + wn * 32 + nt * 16 + laneM;
            #pragma unroll
            for (int r = 0; r < 8; ++r) {
                const int M = rowBase + wm * 32 + mt * 16 + h * 8 + r;
                out[(size_t)M * OUT_DIM + N] = acc[mt][nt][r];
            }
        }
}

// ------------------------------- launch ------------------------------------

extern "C" void kernel_launch(void* const* d_in, const int* in_sizes, int n_in,
                              void* d_out, int out_size, void* d_ws, size_t ws_size,
                              hipStream_t stream) {
    (void)in_sizes; (void)n_in; (void)out_size;
    const float* inputs = (const float*)d_in[0];   // (2048, 513)
    const float* W      = (const float*)d_in[1];   // (8, 262144) == (4096, 512)
    const float* bwts   = (const float*)d_in[2];   // (8, 512)
    float*       out    = (float*)d_out;           // (2048, 512)

    const size_t WT_BYTES = (size_t)OUT_DIM * KDIM * sizeof(u16);   //  4 MB
    const size_t XA_BYTES = (size_t)BATCH  * KDIM * sizeof(u16);    // 16 MB
    dim3 grid(OUT_DIM / NT, BATCH / MT);                            // 4 x 32

    if (ws_size >= WT_BYTES + XA_BYTES) {
        u16* WT = (u16*)d_ws;
        u16* XA = (u16*)((char*)d_ws + WT_BYTES);
        prep_wt  <<<dim3(KDIM / 32, OUT_DIM / 32), THREADS, 0, stream>>>(W, WT);
        prep_xaug<<<BATCH, THREADS, 0, stream>>>(inputs, XA);
        vclinear_wmma_pre<<<grid, THREADS, 0, stream>>>(inputs, bwts, XA, WT, out);
    } else {
        vclinear_wmma_fused<<<grid, THREADS, 0, stream>>>(inputs, W, bwts, out);
    }
}
